// MultiheadedSelfAttentionLayer_1760936591673
// MI455X (gfx1250) — matrix operand order repaired and checked by reading, not compiled, not run on hardware
//
#include <hip/hip_runtime.h>
#include <hip/hip_bf16.h>

#define NN   50000
#define NE_  800000
#define DD   128
#define EDIM 64
#define HH   8

typedef __attribute__((ext_vector_type(16))) _Float16 v16h;
typedef __attribute__((ext_vector_type(8)))  _Float16 v8h;
typedef __attribute__((ext_vector_type(8)))  float    v8f;
typedef __attribute__((ext_vector_type(4)))  float    v4f;

// ---------------------------------------------------------------------------
// Float atomic max via monotonic int mapping (works for all finite floats,
// init must be -inf).
// ---------------------------------------------------------------------------
__device__ __forceinline__ void atomicMaxFloat(float* addr, float val) {
    if (val >= 0.0f) atomicMax((int*)addr, __float_as_int(val));
    else             atomicMin((unsigned int*)addr, __float_as_uint(val));
}

// ---------------------------------------------------------------------------
// Convert a 128x128 f32 weight (row-major, W[k][n]) into WMMA-B-fragment f16
// layout: for kTile t (K=32), colTile c (N=16), lane l, half j:
//   value = W[32t + 8*(l>=16) + (j<8 ? j : 8+j)][16c + (l&15)]
// stored contiguously so each lane reads one aligned 32B v16h per fragment.
// ---------------------------------------------------------------------------
__global__ void wswizzle_kernel(const float* __restrict__ W,
                                _Float16* __restrict__ WH) {
    for (int o = blockIdx.x * blockDim.x + threadIdx.x; o < DD * DD;
         o += blockDim.x * gridDim.x) {
        int j = o & 15;
        int l = (o >> 4) & 31;
        int c = (o >> 9) & 7;
        int t = o >> 12;
        int k = 32 * t + ((l >= 16) ? 8 : 0) + ((j < 8) ? j : 8 + j);
        int n = 16 * c + (l & 15);
        WH[o] = (_Float16)W[k * DD + n];
    }
}

__global__ void fill_kernel(float* __restrict__ p, float v, int n) {
    int i = blockIdx.x * blockDim.x + threadIdx.x;
    if (i < n) p[i] = v;
}

// ---------------------------------------------------------------------------
// Node projections: Qn = V@Wq+bq, Kn = V@Wk+bk, Vn = V@Wv+bv.
// Block = 16-row tile x 128 cols, 8 waves; wave w owns 16-col tile w.
// A tile staged in LDS as f16; A fragments reused across the 3 B matrices.
// ---------------------------------------------------------------------------
__global__ __launch_bounds__(256)
void proj_gemm_kernel(const float* __restrict__ V,
                      const _Float16* __restrict__ WqH,
                      const _Float16* __restrict__ WkH,
                      const _Float16* __restrict__ WvH,
                      const float* __restrict__ bq,
                      const float* __restrict__ bk,
                      const float* __restrict__ bv,
                      float* __restrict__ Qn,
                      float* __restrict__ Kn,
                      float* __restrict__ Vn) {
    __shared__ _Float16 As[16][DD];
    const int tid  = threadIdx.x;
    const int row0 = blockIdx.x * 16;

    // Cooperative load of 16x128 f32 tile -> f16 LDS (each thread: 8 floats)
    {
        int r  = tid >> 4;
        int cc = (tid & 15) * 8;
        const float4* src = (const float4*)(V + (size_t)(row0 + r) * DD + cc);
        float4 a0 = src[0], a1 = src[1];
        v8h h;
        h[0] = (_Float16)a0.x; h[1] = (_Float16)a0.y;
        h[2] = (_Float16)a0.z; h[3] = (_Float16)a0.w;
        h[4] = (_Float16)a1.x; h[5] = (_Float16)a1.y;
        h[6] = (_Float16)a1.z; h[7] = (_Float16)a1.w;
        *(v8h*)&As[r][cc] = h;
    }
    __syncthreads();

    const int lane = tid & 31;
    const int w    = tid >> 5;        // column tile 0..7
    const int m    = lane & 15;       // A row
    const int hi   = lane >> 4;
    const int koff = hi * 8;

    // A fragments per ISA 16-bit 16x32 layout: lane<16: K {0..7,16..23},
    // lane>=16: K {8..15,24..31} (per 32-K tile)
    v16h a[4];
    #pragma unroll
    for (int t = 0; t < 4; ++t) {
        v8h lo = *(const v8h*)&As[m][32 * t + koff];
        v8h hv = *(const v8h*)&As[m][32 * t + 16 + koff];
        a[t] = __builtin_shufflevector(lo, hv, 0, 1, 2, 3, 4, 5, 6, 7,
                                       8, 9, 10, 11, 12, 13, 14, 15);
    }

    v8f cq = {}, ck = {}, cv = {};
    #pragma unroll
    for (int t = 0; t < 4; ++t) {
        const int fo = ((t * 8 + w) * 32 + lane) * 16;
        v16h bqf = *(const v16h*)(WqH + fo);
        v16h bkf = *(const v16h*)(WkH + fo);
        v16h bvf = *(const v16h*)(WvH + fo);
        cq = __builtin_amdgcn_wmma_f32_16x16x32_f16(false, a[t], false, bqf,
                                                    (short)0, cq, false, false);
        ck = __builtin_amdgcn_wmma_f32_16x16x32_f16(false, a[t], false, bkf,
                                                    (short)0, ck, false, false);
        cv = __builtin_amdgcn_wmma_f32_16x16x32_f16(false, a[t], false, bvf,
                                                    (short)0, cv, false, false);
    }

    // C/D layout: vgpr r -> row = r + 8*hi, col = lane&15 (same col all r)
    const int gcol = w * 16 + m;
    const float addq = bq[gcol], addk = bk[gcol], addv = bv[gcol];
    #pragma unroll
    for (int r = 0; r < 8; ++r) {
        int grow = row0 + r + 8 * hi;
        size_t o = (size_t)grow * DD + gcol;
        Qn[o] = cq[r] + addq;
        Kn[o] = ck[r] + addk;
        Vn[o] = cv[r] + addv;
    }
}

// ---------------------------------------------------------------------------
// Per-edge scores: s[e][h] = dot(Qn[src],Kn[dest])_h / 4 + (E[e]@Wb + bb),
// plus running atomic segment-max into m[dest][h].
// Qn/Kn gathers hit L2 (77MB resident in 192MB L2); E is a 205MB one-shot
// stream -> non-temporal loads so it does not evict the gather working set.
// ---------------------------------------------------------------------------
__global__ __launch_bounds__(256)
void edge_scores_kernel(const float* __restrict__ Qn,
                        const float* __restrict__ Kn,
                        const float* __restrict__ E,
                        const float* __restrict__ Wb,
                        const float* __restrict__ bb,
                        const int* __restrict__ src,
                        const int* __restrict__ dst,
                        float* __restrict__ scores,
                        float* __restrict__ mbuf) {
    int e = blockIdx.x * blockDim.x + threadIdx.x;
    if (e >= NE_) return;
    int s = src[e], d = dst[e];

    float bias = bb[0];
    const v4f* Ee = (const v4f*)(E + (size_t)e * EDIM);
    #pragma unroll
    for (int j = 0; j < EDIM / 4; ++j) {
        v4f ev = __builtin_nontemporal_load(Ee + j);   // TH=NT stream
        bias += ev.x * Wb[4 * j + 0] + ev.y * Wb[4 * j + 1] +
                ev.z * Wb[4 * j + 2] + ev.w * Wb[4 * j + 3];
    }

    const v4f* q = (const v4f*)(Qn + (size_t)s * DD);
    const v4f* k = (const v4f*)(Kn + (size_t)d * DD);
    float sc[HH];
    #pragma unroll
    for (int h = 0; h < HH; ++h) sc[h] = 0.0f;
    #pragma unroll
    for (int j = 0; j < DD / 4; ++j) {
        v4f qv = q[j], kv = k[j];
        sc[j >> 2] += qv.x * kv.x + qv.y * kv.y + qv.z * kv.z + qv.w * kv.w;
    }

    v4f s0, s1;
    #pragma unroll
    for (int h = 0; h < 4; ++h) s0[h] = sc[h] * 0.25f + bias;      // 1/sqrt(16)
    #pragma unroll
    for (int h = 0; h < 4; ++h) s1[h] = sc[h + 4] * 0.25f + bias;
    v4f* so = (v4f*)(scores + (size_t)e * HH);
    so[0] = s0;                       // two b128 stores, 32B aligned
    so[1] = s1;
    #pragma unroll
    for (int h = 0; h < 4; ++h) {
        atomicMaxFloat(&mbuf[(size_t)d * HH + h],     s0[h]);
        atomicMaxFloat(&mbuf[(size_t)d * HH + h + 4], s1[h]);
    }
}

// exp(score - m[dest]) in place + atomic segment-sum into den
__global__ __launch_bounds__(256)
void edge_exp_kernel(const int* __restrict__ dst,
                     const float* __restrict__ mbuf,
                     float* __restrict__ scores,
                     float* __restrict__ den) {
    int e = blockIdx.x * blockDim.x + threadIdx.x;
    if (e >= NE_) return;
    int d = dst[e];
    const v4f* mrow = (const v4f*)(mbuf + (size_t)d * HH);
    v4f m0 = mrow[0], m1 = mrow[1];
    v4f* so = (v4f*)(scores + (size_t)e * HH);
    v4f e0 = so[0], e1 = so[1];
    #pragma unroll
    for (int h = 0; h < 4; ++h) e0[h] = __expf(e0[h] - m0[h]);
    #pragma unroll
    for (int h = 0; h < 4; ++h) e1[h] = __expf(e1[h] - m1[h]);
    so[0] = e0;
    so[1] = e1;
    #pragma unroll
    for (int h = 0; h < 4; ++h) {
        atomicAdd(&den[(size_t)d * HH + h],     e0[h]);
        atomicAdd(&den[(size_t)d * HH + h + 4], e1[h]);
    }
}

// alpha sums: S[d][h] = sum_e ex/den  (exact: Vv depends only on dest, so
// Hagg[d] = S[d][h] * Vn[d] — no 128-wide scatter needed)
__global__ __launch_bounds__(256)
void edge_alpha_kernel(const int* __restrict__ dst,
                       const float* __restrict__ den,
                       const float* __restrict__ scores,
                       float* __restrict__ S) {
    int e = blockIdx.x * blockDim.x + threadIdx.x;
    if (e >= NE_) return;
    int d = dst[e];
    const v4f* drow = (const v4f*)(den + (size_t)d * HH);
    const v4f* so   = (const v4f*)(scores + (size_t)e * HH);
    v4f d0 = drow[0], d1 = drow[1];
    v4f e0 = so[0],   e1 = so[1];
    #pragma unroll
    for (int h = 0; h < 4; ++h) {
        atomicAdd(&S[(size_t)d * HH + h],     e0[h] / d0[h]);
        atomicAdd(&S[(size_t)d * HH + h + 4], e1[h] / d1[h]);
    }
}

// ---------------------------------------------------------------------------
// Output: O = (S_head ⊙ Vn) @ Wo + bo, same WMMA tiling as projections.
// O is written once and never re-read -> non-temporal stores.
// ---------------------------------------------------------------------------
__global__ __launch_bounds__(256)
void out_gemm_kernel(const float* __restrict__ Vn,
                     const float* __restrict__ S,
                     const _Float16* __restrict__ WoH,
                     const float* __restrict__ bo,
                     float* __restrict__ O) {
    __shared__ _Float16 As[16][DD];
    const int tid  = threadIdx.x;
    const int row0 = blockIdx.x * 16;
    {
        int r    = tid >> 4;
        int cc   = (tid & 15) * 8;        // 8-aligned => single head per chunk
        int grow = row0 + r;
        float scale = S[(size_t)grow * HH + (cc >> 4)];
        const float4* src = (const float4*)(Vn + (size_t)grow * DD + cc);
        float4 a0 = src[0], a1 = src[1];
        v8h h;
        h[0] = (_Float16)(a0.x * scale); h[1] = (_Float16)(a0.y * scale);
        h[2] = (_Float16)(a0.z * scale); h[3] = (_Float16)(a0.w * scale);
        h[4] = (_Float16)(a1.x * scale); h[5] = (_Float16)(a1.y * scale);
        h[6] = (_Float16)(a1.z * scale); h[7] = (_Float16)(a1.w * scale);
        *(v8h*)&As[r][cc] = h;
    }
    __syncthreads();

    const int lane = tid & 31;
    const int w    = tid >> 5;
    const int m    = lane & 15;
    const int hi   = lane >> 4;
    const int koff = hi * 8;

    v8f acc = {};
    #pragma unroll
    for (int t = 0; t < 4; ++t) {
        v8h lo = *(const v8h*)&As[m][32 * t + koff];
        v8h hv = *(const v8h*)&As[m][32 * t + 16 + koff];
        v16h a = __builtin_shufflevector(lo, hv, 0, 1, 2, 3, 4, 5, 6, 7,
                                         8, 9, 10, 11, 12, 13, 14, 15);
        const int fo = ((t * 8 + w) * 32 + lane) * 16;
        v16h b = *(const v16h*)(WoH + fo);
        acc = __builtin_amdgcn_wmma_f32_16x16x32_f16(false, a, false, b,
                                                     (short)0, acc, false, false);
    }

    const int gcol = w * 16 + m;
    const float add = bo[gcol];
    #pragma unroll
    for (int r = 0; r < 8; ++r) {
        int grow = row0 + r + 8 * hi;
        __builtin_nontemporal_store(acc[r] + add, &O[(size_t)grow * DD + gcol]);
    }
}

// ---------------------------------------------------------------------------
extern "C" void kernel_launch(void* const* d_in, const int* in_sizes, int n_in,
                              void* d_out, int out_size, void* d_ws, size_t ws_size,
                              hipStream_t stream) {
    (void)in_sizes; (void)n_in; (void)out_size; (void)ws_size;
    const float* V  = (const float*)d_in[0];
    const float* E  = (const float*)d_in[1];
    const int*   ei = (const int*)  d_in[2];
    const float* Wq = (const float*)d_in[3];  const float* bq = (const float*)d_in[4];
    const float* Wk = (const float*)d_in[5];  const float* bk = (const float*)d_in[6];
    const float* Wv = (const float*)d_in[7];  const float* bv = (const float*)d_in[8];
    const float* Wb = (const float*)d_in[9];  const float* bb = (const float*)d_in[10];
    const float* Wo = (const float*)d_in[11]; const float* bo = (const float*)d_in[12];
    float* O = (float*)d_out;

    const int* src = ei;
    const int* dst = ei + NE_;

    // Workspace carve-out
    char* ws = (char*)d_ws;
    size_t off = 0;
    _Float16* WqH = (_Float16*)(ws + off); off += (size_t)DD * DD * 2;
    _Float16* WkH = (_Float16*)(ws + off); off += (size_t)DD * DD * 2;
    _Float16* WvH = (_Float16*)(ws + off); off += (size_t)DD * DD * 2;
    _Float16* WoH = (_Float16*)(ws + off); off += (size_t)DD * DD * 2;
    float* Qn     = (float*)(ws + off);    off += (size_t)NN * DD * 4;
    float* Kn     = (float*)(ws + off);    off += (size_t)NN * DD * 4;
    float* Vn     = (float*)(ws + off);    off += (size_t)NN * DD * 4;
    float* scores = (float*)(ws + off);    off += (size_t)NE_ * HH * 4;
    float* mbuf   = (float*)(ws + off);    off += (size_t)NN * HH * 4;
    float* den    = (float*)(ws + off);    off += (size_t)NN * HH * 4;
    float* Ssum   = (float*)(ws + off);    off += (size_t)NN * HH * 4;

    // 1) Weight swizzle f32 -> WMMA-fragment f16
    wswizzle_kernel<<<16, 256, 0, stream>>>(Wq, WqH);
    wswizzle_kernel<<<16, 256, 0, stream>>>(Wk, WkH);
    wswizzle_kernel<<<16, 256, 0, stream>>>(Wv, WvH);
    wswizzle_kernel<<<16, 256, 0, stream>>>(Wo, WoH);

    // 2) Init segment buffers
    const int NH = NN * HH;
    fill_kernel<<<(NH + 255) / 256, 256, 0, stream>>>(mbuf, -__builtin_inff(), NH);
    fill_kernel<<<(NH + 255) / 256, 256, 0, stream>>>(den, 0.0f, NH);
    fill_kernel<<<(NH + 255) / 256, 256, 0, stream>>>(Ssum, 0.0f, NH);

    // 3) Per-node projections (WMMA)
    proj_gemm_kernel<<<NN / 16, 256, 0, stream>>>(V, WqH, WkH, WvH,
                                                  bq, bk, bv, Qn, Kn, Vn);

    // 4) Edge scores + bias + segment max
    edge_scores_kernel<<<NE_ / 256, 256, 0, stream>>>(Qn, Kn, E, Wb, bb,
                                                      src, dst, scores, mbuf);

    // 5) exp + segment sum
    edge_exp_kernel<<<NE_ / 256, 256, 0, stream>>>(dst, mbuf, scores, den);

    // 6) alpha sums per (node, head)
    edge_alpha_kernel<<<NE_ / 256, 256, 0, stream>>>(dst, den, scores, Ssum);

    // 7) Output projection (WMMA) with per-head scaling + bias
    out_gemm_kernel<<<NN / 16, 256, 0, stream>>>(Vn, Ssum, WoH, bo, O);
}
